// AttentionBlock_1769526526709
// MI455X (gfx1250) — matrix operand order repaired and checked by reading, not compiled, not run on hardware
//
#include <hip/hip_runtime.h>
#include <hip/hip_bf16.h>

// ---------------- problem constants ----------------
#define B_      8
#define C_      512
#define L_      1024      // H*W = 32*32
#define NH_     8
#define G_      32
#define CH_     64        // head dim = C/NH
#define ENC_C_  768
#define ENC_L_  77
#define ENC_LP  80        // padded to multiple of 16
#define S_      1101      // ENC_L + L
#define SP_     1120      // padded to multiple of 32 (35 k-steps)
#define TM_     32        // query rows per attention workgroup

typedef __attribute__((ext_vector_type(16))) _Float16 v16h;
typedef __attribute__((ext_vector_type(8)))  float    v8f;

// ---------------- WMMA fragment helpers (wave32, 16x16x32 f16) ----------------
// A fragment: 16x32 tile, row-major with leading dim `ld` (halfs).
// lane<16 : row=lane,    K = 0..15
// lane>=16: row=lane-16, K = 16..31
__device__ __forceinline__ v16h load_a16(const _Float16* __restrict__ p, int ld) {
    int lane = threadIdx.x & 31;
    int row  = lane & 15;
    int kb   = (lane >> 4) * 16;
    const _Float16* r0 = p + (size_t)row * ld + kb;
    v16h f;
#pragma unroll
    for (int j = 0; j < 16; ++j) f[j] = r0[j];
    return f;
}

// B fragment: 32x16 tile given as B^T row-major (rows = N, contiguous K), ld in halfs.
// lane<16 : col=lane,    K = 0..15 ;  lane>=16: col=lane-16, K = 16..31
__device__ __forceinline__ v16h load_b16(const _Float16* __restrict__ p, int ld) {
    int lane = threadIdx.x & 31;
    int col  = lane & 15;
    int kb   = (lane >> 4) * 16;
    const _Float16* r0 = p + (size_t)col * ld + kb;
    v16h f;
#pragma unroll
    for (int j = 0; j < 16; ++j) f[j] = r0[j];
    return f;
}

__device__ __forceinline__ v8f wmma16(v16h a, v16h b, v8f c) {
    return __builtin_amdgcn_wmma_f32_16x16x32_f16(false, a, false, b, (short)0, c, false, false);
}

// ---------------- prep kernels ----------------
__global__ void f32_to_f16_kernel(const float* __restrict__ src, _Float16* __restrict__ dst, int n) {
    int i = blockIdx.x * blockDim.x + threadIdx.x;
    if (i < n) dst[i] = (_Float16)src[i];
}

// encoder_out (B, 768, 77) -> encT (B, 80, 768) f16, zero-padded rows 77..79
__global__ void enc_t_kernel(const float* __restrict__ enc, _Float16* __restrict__ encT) {
    int idx = blockIdx.x * blockDim.x + threadIdx.x;
    const int total = B_ * ENC_LP * ENC_C_;
    if (idx >= total) return;
    int c = idx % ENC_C_;
    int l = (idx / ENC_C_) % ENC_LP;
    int b = idx / (ENC_C_ * ENC_LP);
    float v = (l < ENC_L_) ? enc[((size_t)b * ENC_C_ + c) * ENC_L_ + l] : 0.0f;
    encT[idx] = (_Float16)v;
}

// zero padded S rows of kT / padded S cols of vM (s in [S_, SP_))
__global__ void pad_zero_kernel(_Float16* __restrict__ kT, _Float16* __restrict__ vM) {
    const int NP = SP_ - S_;
    const int total = B_ * NH_ * NP * CH_;
    int idx = blockIdx.x * blockDim.x + threadIdx.x;
    if (idx >= total) return;
    int c  = idx % CH_;
    int sp = (idx / CH_) % NP;
    int bh = idx / (CH_ * NP);
    int s  = S_ + sp;
    kT[((size_t)bh * SP_ + s) * CH_ + c] = (_Float16)0.0f;
    vM[((size_t)bh * CH_ + c) * SP_ + s] = (_Float16)0.0f;
}

// ---------------- GroupNorm: x (B,C,L) f32 -> hnT (B,L,C) f16 ----------------
__global__ __launch_bounds__(256) void gn_kernel(const float* __restrict__ x,
                                                 const float* __restrict__ gw,
                                                 const float* __restrict__ gb,
                                                 _Float16* __restrict__ hnT) {
    const int CPG = C_ / G_;               // 16 channels per group
    const int NE  = CPG * L_;              // 16384 elements per (b,g)
    int b = blockIdx.x / G_;
    int g = blockIdx.x % G_;
    const float* xp = x + ((size_t)b * C_ + (size_t)g * CPG) * L_;

    __shared__ float ssum[256];
    __shared__ float ssq[256];
    float s = 0.f, ss = 0.f;
    for (int e = threadIdx.x; e < NE; e += 256) {
        float v = xp[e];
        s += v; ss += v * v;
    }
    ssum[threadIdx.x] = s; ssq[threadIdx.x] = ss;
    __syncthreads();
    for (int off = 128; off > 0; off >>= 1) {
        if (threadIdx.x < off) {
            ssum[threadIdx.x] += ssum[threadIdx.x + off];
            ssq[threadIdx.x]  += ssq[threadIdx.x + off];
        }
        __syncthreads();
    }
    float mean = ssum[0] * (1.0f / NE);
    float var  = ssq[0] * (1.0f / NE) - mean * mean;
    float rstd = rsqrtf(var + 1e-5f);

    for (int e = threadIdx.x; e < NE; e += 256) {
        int cl = e >> 10;                  // /L_
        int l  = e & (L_ - 1);
        int c  = g * CPG + cl;
        float v = (xp[e] - mean) * rstd * gw[c] + gb[c];
        hnT[((size_t)b * L_ + l) * C_ + c] = (_Float16)v;
    }
}

// ---------------- QKV GEMM: (1536x512) x (512x1024) per batch ----------------
// grid (24, 4, 8), block 128 (4 waves). Wave tile 64x64, block tile 64x256.
__global__ __launch_bounds__(128) void qkv_kernel(const _Float16* __restrict__ W,
                                                  const float* __restrict__ bias,
                                                  const _Float16* __restrict__ hnT,
                                                  _Float16* __restrict__ qT,
                                                  _Float16* __restrict__ kT,
                                                  _Float16* __restrict__ vM) {
    int b    = blockIdx.z;
    int m0   = blockIdx.x * 64;
    int wv   = threadIdx.x >> 5;
    int n0   = blockIdx.y * 256 + wv * 64;
    int lane = threadIdx.x & 31;
    const _Float16* Bp = hnT + (size_t)b * L_ * C_;

    v8f acc[4][4] = {};
    for (int k0 = 0; k0 < C_; k0 += 32) {
        v16h af[4], bf[4];
#pragma unroll
        for (int mi = 0; mi < 4; ++mi)
            af[mi] = load_a16(W + (size_t)(m0 + mi * 16) * C_ + k0, C_);
#pragma unroll
        for (int ni = 0; ni < 4; ++ni)
            bf[ni] = load_b16(Bp + (size_t)(n0 + ni * 16) * C_ + k0, C_);
#pragma unroll
        for (int mi = 0; mi < 4; ++mi)
#pragma unroll
            for (int ni = 0; ni < 4; ++ni)
                acc[mi][ni] = wmma16(af[mi], bf[ni], acc[mi][ni]);
    }

    int mo = (lane >> 4) * 8;
    int nn = lane & 15;
#pragma unroll
    for (int mi = 0; mi < 4; ++mi) {
#pragma unroll
        for (int ni = 0; ni < 4; ++ni) {
#pragma unroll
            for (int r = 0; r < 8; ++r) {
                int o = m0 + mi * 16 + mo + r;
                int l = n0 + ni * 16 + nn;
                float v = acc[mi][ni][r] + bias[o];
                int head = o / 192, rem = o % 192;
                int part = rem / 64, c = rem % 64;
                int bh = b * NH_ + head;
                _Float16 hv = (_Float16)v;
                if (part == 0)      qT[((size_t)bh * L_ + l) * CH_ + c] = hv;
                else if (part == 1) kT[((size_t)bh * SP_ + (ENC_L_ + l)) * CH_ + c] = hv;
                else                vM[((size_t)bh * CH_ + c) * SP_ + (ENC_L_ + l)] = hv;
            }
        }
    }
}

// ---------------- EKV GEMM: (1024x768) x (768x80) per batch ----------------
// grid (16, 5, 8), block 32 (1 wave). Wave tile 64x16.
__global__ __launch_bounds__(32) void ekv_kernel(const _Float16* __restrict__ W,
                                                 const float* __restrict__ bias,
                                                 const _Float16* __restrict__ encT,
                                                 _Float16* __restrict__ kT,
                                                 _Float16* __restrict__ vM) {
    int b    = blockIdx.z;
    int m0   = blockIdx.x * 64;
    int n0   = blockIdx.y * 16;
    int lane = threadIdx.x & 31;
    const _Float16* Bp = encT + (size_t)b * ENC_LP * ENC_C_;

    v8f acc[4] = {};
    for (int k0 = 0; k0 < ENC_C_; k0 += 32) {
        v16h bf = load_b16(Bp + (size_t)n0 * ENC_C_ + k0, ENC_C_);
#pragma unroll
        for (int mi = 0; mi < 4; ++mi) {
            v16h af = load_a16(W + (size_t)(m0 + mi * 16) * ENC_C_ + k0, ENC_C_);
            acc[mi] = wmma16(af, bf, acc[mi]);
        }
    }

    int mo = (lane >> 4) * 8;
    int nn = lane & 15;
    int l  = n0 + nn;
    if (l < ENC_L_) {
#pragma unroll
        for (int mi = 0; mi < 4; ++mi)
#pragma unroll
            for (int r = 0; r < 8; ++r) {
                int o = m0 + mi * 16 + mo + r;
                float v = acc[mi][r] + bias[o];
                int head = o / 128, rem = o % 128;
                int part = rem / 64, c = rem % 64;
                int bh = b * NH_ + head;
                _Float16 hv = (_Float16)v;
                if (part == 0) kT[((size_t)bh * SP_ + l) * CH_ + c] = hv;
                else           vM[((size_t)bh * CH_ + c) * SP_ + l] = hv;
            }
    }
}

// ---------------- fused attention per (head, 32-row query tile) ----------------
// grid (64, 32), block 128 (4 waves). dynamic LDS: scores f32 + probs f16 + q tile.
__global__ __launch_bounds__(128) void attn_kernel(const _Float16* __restrict__ qT,
                                                   const _Float16* __restrict__ kT,
                                                   const _Float16* __restrict__ vM,
                                                   _Float16* __restrict__ aT) {
    extern __shared__ char smem[];
    float*    sc = (float*)smem;                       // [TM_][SP_]
    _Float16* pr = (_Float16*)(sc + TM_ * SP_);        // [TM_][SP_]
    _Float16* qs = pr + TM_ * SP_;                     // [TM_][CH_]

    int bh = blockIdx.x;
    int t0 = blockIdx.y * TM_;
    int b  = bh / NH_, nh = bh % NH_;
    int tid = threadIdx.x, lane = tid & 31, wv = tid >> 5;

    // stage q tile to LDS
    {
        const _Float16* qsrc = qT + ((size_t)bh * L_ + t0) * CH_;
        for (int e = tid; e < TM_ * CH_; e += 128) qs[e] = qsrc[e];
    }
    __syncthreads();

    // ---- phase 1: scores = 0.125 * q^T k   (s-tiles distributed over waves)
    const int NST = SP_ / 16;                          // 70
    for (int st = wv; st < NST; st += 4) {
        v8f acc[2] = {};
        for (int k0 = 0; k0 < CH_; k0 += 32) {
            v16h bf = load_b16(kT + ((size_t)bh * SP_ + st * 16) * CH_ + k0, CH_);
#pragma unroll
            for (int mi = 0; mi < 2; ++mi) {
                v16h af = load_a16(qs + mi * 16 * CH_ + k0, CH_);
                acc[mi] = wmma16(af, bf, acc[mi]);
            }
        }
        int mo = (lane >> 4) * 8, nn = lane & 15;
#pragma unroll
        for (int mi = 0; mi < 2; ++mi)
#pragma unroll
            for (int r = 0; r < 8; ++r)
                sc[(mi * 16 + mo + r) * SP_ + st * 16 + nn] = acc[mi][r] * 0.125f;
    }
    __syncthreads();

    // ---- phase 2: row softmax (8 rows per wave), masked beyond S_
    for (int row = wv * (TM_ / 4); row < (wv + 1) * (TM_ / 4); ++row) {
        float* srow = sc + (size_t)row * SP_;
        float m = -1e30f;
        for (int s = lane; s < S_; s += 32) m = fmaxf(m, srow[s]);
#pragma unroll
        for (int off = 16; off > 0; off >>= 1) m = fmaxf(m, __shfl_xor(m, off, 32));
        float sum = 0.f;
        for (int s = lane; s < SP_; s += 32) {
            float p = (s < S_) ? __expf(srow[s] - m) : 0.0f;
            srow[s] = p;
            sum += p;
        }
#pragma unroll
        for (int off = 16; off > 0; off >>= 1) sum += __shfl_xor(sum, off, 32);
        float inv = 1.0f / sum;
        _Float16* prow = pr + (size_t)row * SP_;
        for (int s = lane; s < SP_; s += 32) prow[s] = (_Float16)(srow[s] * inv);
    }
    __syncthreads();

    // ---- phase 3: out = P (TM_ x SP_) x V^T (SP_ x 64); 16 output cols per wave
    {
        int c0 = wv * 16;
        v8f acc[2] = {};
        for (int k0 = 0; k0 < SP_; k0 += 32) {
            v16h bf = load_b16(vM + ((size_t)bh * CH_ + c0) * SP_ + k0, SP_);
#pragma unroll
            for (int mi = 0; mi < 2; ++mi) {
                v16h af = load_a16(pr + (size_t)(mi * 16) * SP_ + k0, SP_);
                acc[mi] = wmma16(af, bf, acc[mi]);
            }
        }
        int mo = (lane >> 4) * 8, nn = lane & 15;
#pragma unroll
        for (int mi = 0; mi < 2; ++mi)
#pragma unroll
            for (int r = 0; r < 8; ++r) {
                int t  = t0 + mi * 16 + mo + r;
                int cg = nh * CH_ + c0 + nn;
                aT[((size_t)b * L_ + t) * C_ + cg] = (_Float16)acc[mi][r];
            }
    }
}

// ---------------- projection + bias + residual ----------------
// grid (8, 4, 8), block 128 (4 waves), wave tile 64x64.
__global__ __launch_bounds__(128) void proj_kernel(const _Float16* __restrict__ W,
                                                   const float* __restrict__ bias,
                                                   const _Float16* __restrict__ aT,
                                                   const float* __restrict__ x,
                                                   float* __restrict__ out) {
    int b    = blockIdx.z;
    int m0   = blockIdx.x * 64;
    int wv   = threadIdx.x >> 5;
    int n0   = blockIdx.y * 256 + wv * 64;
    int lane = threadIdx.x & 31;
    const _Float16* Bp = aT + (size_t)b * L_ * C_;

    v8f acc[4][4] = {};
    for (int k0 = 0; k0 < C_; k0 += 32) {
        v16h af[4], bf[4];
#pragma unroll
        for (int mi = 0; mi < 4; ++mi)
            af[mi] = load_a16(W + (size_t)(m0 + mi * 16) * C_ + k0, C_);
#pragma unroll
        for (int ni = 0; ni < 4; ++ni)
            bf[ni] = load_b16(Bp + (size_t)(n0 + ni * 16) * C_ + k0, C_);
#pragma unroll
        for (int mi = 0; mi < 4; ++mi)
#pragma unroll
            for (int ni = 0; ni < 4; ++ni)
                acc[mi][ni] = wmma16(af[mi], bf[ni], acc[mi][ni]);
    }

    int mo = (lane >> 4) * 8;
    int nn = lane & 15;
#pragma unroll
    for (int mi = 0; mi < 4; ++mi)
#pragma unroll
        for (int ni = 0; ni < 4; ++ni)
#pragma unroll
            for (int r = 0; r < 8; ++r) {
                int o = m0 + mi * 16 + mo + r;
                int l = n0 + ni * 16 + nn;
                size_t idx = ((size_t)b * C_ + o) * L_ + l;
                out[idx] = x[idx] + bias[o] + acc[mi][ni][r];
            }
}

// ---------------- launcher ----------------
extern "C" void kernel_launch(void* const* d_in, const int* in_sizes, int n_in,
                              void* d_out, int out_size, void* d_ws, size_t ws_size,
                              hipStream_t stream) {
    (void)in_sizes; (void)n_in; (void)out_size; (void)ws_size;
    const float* x     = (const float*)d_in[0];
    const float* enc   = (const float*)d_in[1];
    const float* gnw   = (const float*)d_in[2];
    const float* gnb   = (const float*)d_in[3];
    const float* qkvw  = (const float*)d_in[4];
    const float* qkvb  = (const float*)d_in[5];
    const float* ekvw  = (const float*)d_in[6];
    const float* ekvb  = (const float*)d_in[7];
    const float* projw = (const float*)d_in[8];
    const float* projb = (const float*)d_in[9];
    float* out = (float*)d_out;

    // bump allocator over workspace
    char* wsp = (char*)d_ws;
    auto alloc = [&](size_t bytes) -> char* {
        char* p = wsp;
        wsp += (bytes + 255) & ~(size_t)255;
        return p;
    };
    _Float16* qkvW16 = (_Float16*)alloc((size_t)3 * C_ * C_ * 2);          // 1536x512
    _Float16* ekvW16 = (_Float16*)alloc((size_t)2 * C_ * ENC_C_ * 2);      // 1024x768
    _Float16* projW16= (_Float16*)alloc((size_t)C_ * C_ * 2);              // 512x512
    _Float16* hnT    = (_Float16*)alloc((size_t)B_ * L_ * C_ * 2);
    _Float16* encT   = (_Float16*)alloc((size_t)B_ * ENC_LP * ENC_C_ * 2);
    _Float16* qT     = (_Float16*)alloc((size_t)B_ * NH_ * L_ * CH_ * 2);
    _Float16* kT     = (_Float16*)alloc((size_t)B_ * NH_ * SP_ * CH_ * 2);
    _Float16* vM     = (_Float16*)alloc((size_t)B_ * NH_ * CH_ * SP_ * 2);
    _Float16* aT     = (_Float16*)alloc((size_t)B_ * L_ * C_ * 2);

    // prep: weight conversion f32->f16
    {
        int n1 = 3 * C_ * C_;
        int n2 = 2 * C_ * ENC_C_;
        int n3 = C_ * C_;
        f32_to_f16_kernel<<<(n1 + 255) / 256, 256, 0, stream>>>(qkvw,  qkvW16,  n1);
        f32_to_f16_kernel<<<(n2 + 255) / 256, 256, 0, stream>>>(ekvw,  ekvW16,  n2);
        f32_to_f16_kernel<<<(n3 + 255) / 256, 256, 0, stream>>>(projw, projW16, n3);
    }
    // prep: encoder transpose + pad
    {
        int n = B_ * ENC_LP * ENC_C_;
        enc_t_kernel<<<(n + 255) / 256, 256, 0, stream>>>(enc, encT);
    }
    // prep: zero padded S region of kT / vM
    {
        int n = B_ * NH_ * (SP_ - S_) * CH_;
        pad_zero_kernel<<<(n + 255) / 256, 256, 0, stream>>>(kT, vM);
    }
    // GroupNorm
    gn_kernel<<<B_ * G_, 256, 0, stream>>>(x, gnw, gnb, hnT);
    // QKV projection
    {
        dim3 grid(3 * C_ / 64, L_ / 256, B_);
        qkv_kernel<<<grid, 128, 0, stream>>>(qkvW16, qkvb, hnT, qT, kT, vM);
    }
    // encoder KV projection
    {
        dim3 grid(2 * C_ / 64, ENC_LP / 16, B_);
        ekv_kernel<<<grid, 32, 0, stream>>>(ekvW16, ekvb, encT, kT, vM);
    }
    // fused attention
    {
        dim3 grid(B_ * NH_, L_ / TM_);
        size_t smem = (size_t)TM_ * SP_ * 4 + (size_t)TM_ * SP_ * 2 + (size_t)TM_ * CH_ * 2;
        attn_kernel<<<grid, 128, smem, stream>>>(qT, kT, vM, aT);
    }
    // projection + residual
    {
        dim3 grid(C_ / 64, L_ / 256, B_);
        proj_kernel<<<grid, 128, 0, stream>>>(projW16, projb, aT, x, out);
    }
}
